// SentenceAttention_50964081934380
// MI455X (gfx1250) — matrix operand order repaired and checked by reading, not compiled, not run on hardware
//
#include <hip/hip_runtime.h>
#include <hip/hip_bf16.h>
#include <math.h>

#define NB 512
#define NS 128
#define ND 512
#define NA 128
#define EPSF 1e-7f

typedef __attribute__((ext_vector_type(16))) __bf16 v16bf;
typedef __attribute__((ext_vector_type(8)))  float  v8f;

__device__ __forceinline__ __bf16 f2bf(float x) { return (__bf16)x; }

// gfx1250 has a native V_TANH_F32 transcendental op — avoid the branchy libm tanhf.
__device__ __forceinline__ float fast_tanh(float x) {
#if defined(__has_builtin) && __has_builtin(__builtin_amdgcn_tanhf)
    return __builtin_amdgcn_tanhf(x);
#else
    float r;
    asm volatile("v_tanh_f32 %0, %1\n\ts_nop 1" : "=v"(r) : "v"(x));
    return r;
#endif
}

// XOR-butterfly add over the 16 lanes of each wave half, via ds_swizzle_b32
// group-of-32 mode (CDNA5 ISA: offset = {xor[14:10], or[9:5], and[4:0]}).
// SWAPX1/2/4/8 patterns; xor masks < 16 never cross the 16-lane half boundary.
__device__ __forceinline__ float half16_sum(float v) {
    int i = __float_as_int(v);
    i = __float_as_int(__int_as_float(i) +
        __int_as_float(__builtin_amdgcn_ds_swizzle(i, 0x041F)));   // xor 1
    i = __float_as_int(__int_as_float(i) +
        __int_as_float(__builtin_amdgcn_ds_swizzle(i, 0x081F)));   // xor 2
    i = __float_as_int(__int_as_float(i) +
        __int_as_float(__builtin_amdgcn_ds_swizzle(i, 0x101F)));   // xor 4
    i = __float_as_int(__int_as_float(i) +
        __int_as_float(__builtin_amdgcn_ds_swizzle(i, 0x201F)));   // xor 8
    return __int_as_float(i);
}

// -----------------------------------------------------------------------------
// Kernel 1: kv[b,s] = tanh(inputs[b,s,:] @ kernel + bias) @ u_context
// grid = NB*8 (each block: one 16-row stripe of one batch), block = 256 (8 waves)
// Each wave computes one 16x16 output tile with v_wmma_f32_16x16x32_bf16,
// K = ND = 512 -> 16 WMMA steps.
// -----------------------------------------------------------------------------
__global__ void __launch_bounds__(256)
attn_scores_wmma(const float* __restrict__ inp, const float* __restrict__ wk,
                 const float* __restrict__ bias, const float* __restrict__ u,
                 float* __restrict__ kv)
{
    const int b    = blockIdx.x >> 3;
    const int i0   = (blockIdx.x & 7) * 16;   // row stripe (S dimension)
    const int wave = threadIdx.x >> 5;
    const int lane = threadIdx.x & 31;
    const int hlf  = lane >> 4;               // 0 -> lanes 0-15, 1 -> lanes 16-31
    const int l16  = lane & 15;
    const int j0   = wave * 16;               // column tile (A dimension)

    // A fragment (16x32 bf16): lane holds row M = l16; elements 0..7 = K(kb..kb+7),
    // 8..15 = K(kb+16..kb+23), kb = hlf*8   (CDNA5 ISA 16-bit A layout)
    const float* arow = inp + ((size_t)b * NS + (size_t)(i0 + l16)) * ND + hlf * 8;
    // B fragment (32x16 bf16): lane holds column N = j0+l16; elements same K order as A
    const float* brow = wk + (size_t)(hlf * 8) * NA + j0 + l16;

    v8f acc = {};
    for (int k0 = 0; k0 < ND; k0 += 32) {
        v16bf af, bfm;
        #pragma unroll
        for (int e = 0; e < 8; ++e) {
            af[e]     = f2bf(arow[k0 + e]);
            af[8 + e] = f2bf(arow[k0 + 16 + e]);
        }
        const float* bp = brow + (size_t)k0 * NA;
        #pragma unroll
        for (int e = 0; e < 8; ++e) {
            bfm[e]     = f2bf(bp[(size_t)e * NA]);
            bfm[8 + e] = f2bf(bp[(size_t)(16 + e) * NA]);
        }
        acc = __builtin_amdgcn_wmma_f32_16x16x32_bf16(false, af, false, bfm,
                                                      (short)0, acc, false, false);
    }

    // C layout: VGPR r -> (M = r + hlf*8, N = l16). Apply bias+tanh, scale by
    // u_context[N], reduce over N (16 lanes of same half) per row.
    const float bia = bias[j0 + l16];
    const float uc  = u[j0 + l16];

    __shared__ float part[8][16];
    #pragma unroll
    for (int r = 0; r < 8; ++r) {
        float v = half16_sum(fast_tanh(acc[r] + bia) * uc);
        if (l16 == 0) part[wave][hlf * 8 + r] = v;   // partial kv for row
    }
    __syncthreads();
    // Deterministic fixed-order combine across the 8 column tiles.
    if (threadIdx.x < 16) {
        float s = 0.f;
        #pragma unroll
        for (int w = 0; w < 8; ++w) s += part[w][threadIdx.x];
        kv[(size_t)b * NS + i0 + threadIdx.x] = s;
    }
}

// -----------------------------------------------------------------------------
// Kernel 2: total = sum(kv) over all B*S (deterministic single-block tree)
// -----------------------------------------------------------------------------
__global__ void __launch_bounds__(256)
reduce_total(const float* __restrict__ kv, float* __restrict__ total)
{
    __shared__ float red[256];
    float acc = 0.f;
    for (int i = threadIdx.x; i < NB * NS; i += 256) acc += kv[i];
    red[threadIdx.x] = acc;
    __syncthreads();
    for (int off = 128; off > 0; off >>= 1) {
        if ((int)threadIdx.x < off) red[threadIdx.x] += red[threadIdx.x + off];
        __syncthreads();
    }
    if (threadIdx.x == 0) *total = red[0];
}

// -----------------------------------------------------------------------------
// Kernel 3: per batch: softmax-with-global-shift, write prob, pooled GEMV.
// grid = NB, block = 256
// -----------------------------------------------------------------------------
__global__ void __launch_bounds__(256)
finalize(const float* __restrict__ inp, const float* __restrict__ mask,
         const float* __restrict__ kv, const float* __restrict__ total,
         float* __restrict__ pooled, float* __restrict__ prob_out)
{
    const int b   = blockIdx.x;
    const int tid = threadIdx.x;
    __shared__ float prob[NS];
    __shared__ float red[NS];

    const float t = *total;
    if (tid < NS) {
        float e = (__expf(kv[(size_t)b * NS + tid] - t) + EPSF) * mask[(size_t)b * NS + tid];
        prob[tid] = e;
        red[tid]  = e;
    }
    __syncthreads();
    for (int off = NS / 2; off > 0; off >>= 1) {
        if (tid < off) red[tid] += red[tid + off];
        __syncthreads();
    }
    const float inv = 1.0f / (red[0] + EPSF);
    if (tid < NS) {
        prob[tid] *= inv;
        prob_out[(size_t)b * NS + tid] = prob[tid];
    }
    __syncthreads();

    const float* base = inp + (size_t)b * NS * ND;
    for (int d = tid; d < ND; d += 256) {
        float acc = 0.f;
        for (int s = 0; s < NS; ++s)
            acc += base[(size_t)s * ND + d] * prob[s];
        pooled[(size_t)b * ND + d] = acc;
    }
}

// -----------------------------------------------------------------------------
extern "C" void kernel_launch(void* const* d_in, const int* in_sizes, int n_in,
                              void* d_out, int out_size, void* d_ws, size_t ws_size,
                              hipStream_t stream)
{
    (void)in_sizes; (void)n_in; (void)out_size; (void)ws_size;
    const float* inp  = (const float*)d_in[0];   // [B,S,D]
    const float* mask = (const float*)d_in[1];   // [B,S,1]
    const float* wk   = (const float*)d_in[2];   // [D,A]
    const float* bias = (const float*)d_in[3];   // [A]
    const float* u    = (const float*)d_in[4];   // [A,1]

    float* out    = (float*)d_out;
    float* pooled = out;                          // [B,D]
    float* prob   = out + (size_t)NB * ND;        // [B,S]

    float* kv    = (float*)d_ws;                  // [B*S]
    float* total = kv + (size_t)NB * NS;          // scalar

    attn_scores_wmma<<<NB * 8, 256, 0, stream>>>(inp, wk, bias, u, kv);
    reduce_total<<<1, 256, 0, stream>>>(kv, total);
    finalize<<<NB, 256, 0, stream>>>(inp, mask, kv, total, pooled, prob);
}
